// SQQuantLoRALinearW4A4_65859028517019
// MI455X (gfx1250) — compile-verified, matrix operand order checked by reading, not run on hardware
//
#include <hip/hip_runtime.h>

// ---------------------------------------------------------------------------
// SmoothQuant + int4-fake-quant LoRA linear, CDNA5 (gfx1250).
// bf16 WMMA GEMM with TDM (tensor_load_to_lds) double-buffered LDS staging.
// ---------------------------------------------------------------------------

typedef __attribute__((ext_vector_type(16))) __bf16 v16bf;
typedef __attribute__((ext_vector_type(8)))  __bf16 bf16x8;
typedef __attribute__((ext_vector_type(8)))  float  v8f;

#define IN_F   4096
#define OUT_F  4096
#define MROWS  8192      // B*S = 4*2048
#define RANK   8
#define QMAXF  7.0f
#define EPSF   1e-8f
#define SCALING 2.0f     // lora_alpha / rank = 16/8

__device__ __forceinline__ unsigned short f2bf(float f) {
  unsigned int x = __float_as_uint(f);
  unsigned int r = x + 0x7FFFu + ((x >> 16) & 1u);   // round-to-nearest-even
  return (unsigned short)(r >> 16);
}
__device__ __forceinline__ float bf2f(unsigned short u) {
  return __uint_as_float(((unsigned int)u) << 16);
}

// ---------------------------------------------------------------------------
// TDM support (gated: falls back to manual staging if builtin is absent)
// ---------------------------------------------------------------------------
#if defined(__has_builtin)
#if __has_builtin(__builtin_amdgcn_tensor_load_to_lds) && \
    __has_builtin(__builtin_amdgcn_s_wait_tensorcnt)
#define USE_TDM 1
#endif
#endif
#ifndef USE_TDM
#define USE_TDM 0
#endif

#if USE_TDM
typedef unsigned int v4u __attribute__((ext_vector_type(4)));
typedef int          v8i __attribute__((ext_vector_type(8)));
typedef int          v4i __attribute__((ext_vector_type(4)));

// 2D tile load: tile_k elems per row (bf16), tile_rows rows, row stride in
// elems. LDS rows are padded +16B via the TDM pad engine (72*2B = 144B pitch).
__device__ __forceinline__ void tdm_load_tile(unsigned lds_addr,
                                              const unsigned short* gptr,
                                              unsigned tile_k,
                                              unsigned tile_rows,
                                              unsigned row_stride_elems) {
  unsigned long long ga = (unsigned long long)(uintptr_t)gptr;
  v4u g0;
  g0.x = 1u;                                                // count=1, user desc
  g0.y = lds_addr;                                          // LDS byte address
  g0.z = (unsigned)ga;                                      // global_addr[31:0]
  g0.w = (unsigned)((ga >> 32) & 0x01FFFFFFu) | (2u << 30); // ga[56:32], type=2
  const unsigned td0 = row_stride_elems;                    // tensor_dim0
  const unsigned td1 = 0x00100000u;                         // tensor_dim1 (large)
  v8i g1;
  g1[0] = (int)((1u << 16)      // data_size = 1 -> 2 bytes
              | (1u << 20)      // pad_enable
              | (4u << 22)      // pad_interval code 4 = 32 DWORDs (128B row)
              | (3u << 25));    // pad_amount  code 3 = 4 DWORDs (16B pad)
  g1[1] = (int)((td0 & 0xFFFFu) << 16);                 // td0[15:0] @ bits 63:48
  g1[2] = (int)((td0 >> 16) | ((td1 & 0xFFFFu) << 16)); // td0[31:16], td1[15:0]
  g1[3] = (int)((td1 >> 16) | (tile_k << 16));          // td1[31:16], tile_dim0
  g1[4] = (int)(tile_rows & 0xFFFFu);                   // tile_dim1, tile_dim2=0
  g1[5] = (int)row_stride_elems;                        // tensor_dim0_stride lo
  g1[6] = 0;
  g1[7] = 0;
  v4i gz = {0, 0, 0, 0};
#if __clang_major__ >= 23
  v8i gz8 = {0, 0, 0, 0, 0, 0, 0, 0};
  __builtin_amdgcn_tensor_load_to_lds(g0, g1, gz, gz, gz8, 0);
#else
  __builtin_amdgcn_tensor_load_to_lds(g0, g1, gz, gz, 0);
#endif
}
#endif  // USE_TDM

// ---------------------------------------------------------------------------
// Kernel 1: per-channel 0.9995 quantile of |x[:,c]| via streaming top-6,
// then amax/s, new_scale, EMA, final activation scale a_vec[c].
// ---------------------------------------------------------------------------
__global__ __launch_bounds__(256)
void quantile_scale_kernel(const float* __restrict__ x,
                           const float* __restrict__ s_vec,
                           const float* __restrict__ a_scale,
                           float* __restrict__ a_vec) {
  const int c = blockIdx.x * blockDim.x + threadIdx.x;
  if (c >= IN_F) return;
  float t0 = 0.f, t1 = 0.f, t2 = 0.f, t3 = 0.f, t4 = 0.f, t5 = 0.f; // |x| >= 0
  const float* xc = x + c;
  for (int row = 0; row < MROWS; ++row) {
    float v = fabsf(xc[(size_t)row * IN_F]);
    if (v > t5) {
      if (v > t0)      { t5 = t4; t4 = t3; t3 = t2; t2 = t1; t1 = t0; t0 = v; }
      else if (v > t1) { t5 = t4; t4 = t3; t3 = t2; t2 = t1; t1 = v; }
      else if (v > t2) { t5 = t4; t4 = t3; t3 = t2; t2 = v; }
      else if (v > t3) { t5 = t4; t4 = t3; t3 = v; }
      else if (v > t4) { t5 = t4; t4 = v; }
      else             { t5 = v; }
    }
  }
  // pos = 0.9995*(8192-1) = 8186.9045 -> interp between sorted idx 8186 (=t5,
  // the 6th largest) and 8187 (=t4, the 5th largest).
  const double pos = 0.9995 * (double)(MROWS - 1);
  const float frac = (float)(pos - (double)(int)pos);
  float q = t5 + frac * (t4 - t5);
  float amax = q / s_vec[c];                 // quantile(|x/s|) = quantile(|x|)/s
  float ns = fmaxf(amax / QMAXF, EPSF);
  float ema = a_scale[c] * 0.99f + ns * 0.01f;
  float a = fmaxf(ema, ns);
  a_vec[c] = fmaxf(a, EPSF);
}

// ---------------------------------------------------------------------------
// Kernel 2: STE fake-quant of activations -> bf16 Xq [MROWS x IN_F]
// ---------------------------------------------------------------------------
__global__ __launch_bounds__(256)
void act_quant_kernel(const float* __restrict__ x,
                      const float* __restrict__ s_vec,
                      const float* __restrict__ a_vec,
                      unsigned short* __restrict__ Xq) {
  size_t idx4 = ((size_t)blockIdx.x * 256 + threadIdx.x) * 4;
  int c = (int)(idx4 & (IN_F - 1));
  float4 xv = *(const float4*)(x + idx4);
  float4 sv = *(const float4*)(s_vec + c);
  float4 av = *(const float4*)(a_vec + c);
  float q0 = rintf(fminf(fmaxf(xv.x / (sv.x * av.x), -QMAXF), QMAXF)) * av.x;
  float q1 = rintf(fminf(fmaxf(xv.y / (sv.y * av.y), -QMAXF), QMAXF)) * av.y;
  float q2 = rintf(fminf(fmaxf(xv.z / (sv.z * av.z), -QMAXF), QMAXF)) * av.z;
  float q3 = rintf(fminf(fmaxf(xv.w / (sv.w * av.w), -QMAXF), QMAXF)) * av.w;
  ushort4 o;
  o.x = f2bf(q0); o.y = f2bf(q1); o.z = f2bf(q2); o.w = f2bf(q3);
  *(ushort4*)(Xq + idx4) = o;
}

// ---------------------------------------------------------------------------
// Kernel 3: weight f32 -> bf16 convert (halves GEMM streaming bytes)
// ---------------------------------------------------------------------------
__global__ __launch_bounds__(256)
void wconv_kernel(const float* __restrict__ w, unsigned short* __restrict__ Wb) {
  size_t idx4 = ((size_t)blockIdx.x * 256 + threadIdx.x) * 4;
  float4 wv = *(const float4*)(w + idx4);
  ushort4 o;
  o.x = f2bf(wv.x); o.y = f2bf(wv.y); o.z = f2bf(wv.z); o.w = f2bf(wv.w);
  *(ushort4*)(Wb + idx4) = o;
}

// ---------------------------------------------------------------------------
// Kernel 4: T = Xq @ lora_A   [MROWS x RANK], one block per row
// ---------------------------------------------------------------------------
__global__ __launch_bounds__(256)
void lora_t_kernel(const unsigned short* __restrict__ Xq,
                   const float* __restrict__ loraA,
                   float* __restrict__ T) {
  __shared__ float red[256 * RANK];
  const int m = blockIdx.x;
  const int tid = threadIdx.x;
  float acc[RANK];
#pragma unroll
  for (int r = 0; r < RANK; ++r) acc[r] = 0.0f;
  const unsigned short* xr = Xq + (size_t)m * IN_F;
  for (int i = tid; i < IN_F; i += 256) {
    float xv = bf2f(xr[i]);
    const float4* ap = (const float4*)(loraA + (size_t)i * RANK);
    float4 a0 = ap[0], a1 = ap[1];
    acc[0] += xv * a0.x; acc[1] += xv * a0.y; acc[2] += xv * a0.z; acc[3] += xv * a0.w;
    acc[4] += xv * a1.x; acc[5] += xv * a1.y; acc[6] += xv * a1.z; acc[7] += xv * a1.w;
  }
#pragma unroll
  for (int r = 0; r < RANK; ++r) red[tid * RANK + r] = acc[r];
  __syncthreads();
  for (int s = 128; s > 0; s >>= 1) {
    if (tid < s) {
#pragma unroll
      for (int r = 0; r < RANK; ++r)
        red[tid * RANK + r] += red[(tid + s) * RANK + r];
    }
    __syncthreads();
  }
  if (tid < RANK) T[(size_t)m * RANK + tid] = red[tid];
}

// ---------------------------------------------------------------------------
// Kernel 5: main bf16 WMMA GEMM + fused LoRA/bias epilogue
//   out[m,n] = sum_k Xq[m,k]*Wb[n,k] + 2*sum_r T[m,r]*loraB[r,n] + bias[n]
// Block tile 128x128, K-chunk 64 staged by TDM into double-buffered LDS,
// 8 waves each computing 32x64 (2x4 tiles, 16 WMMA per chunk).
// ---------------------------------------------------------------------------
#define BM 128
#define BN 128
#define BK 64
#define LDT (BK + 8)   // 72 bf16 elems per LDS row (144B pitch, 16B-aligned)

__device__ __forceinline__ v16bf load_frag(const unsigned short* p) {
  // lanes 0-15: K = {0..7, 16..23}; lanes 16-31: K = {8..15, 24..31}
  // (caller already added half*8 to p)
  union { struct { bf16x8 lo, hi; } s; v16bf v; } u;
  u.s.lo = *(const bf16x8*)(p);
  u.s.hi = *(const bf16x8*)(p + 16);
  return u.v;
}

__global__ __launch_bounds__(256)
void wmma_gemm_kernel(const unsigned short* __restrict__ Xq,
                      const unsigned short* __restrict__ Wb,
                      const float* __restrict__ T,
                      const float* __restrict__ loraB,
                      const float* __restrict__ bias,
                      float* __restrict__ out) {
  const int tid  = threadIdx.x;
  const int lane = tid & 31;            // wave32
  const int wave = tid >> 5;            // 0..7
  const int wrow = wave >> 1;           // 0..3  -> 32-row strip
  const int wcol = wave & 1;            // 0..1  -> 64-col strip
  const int m0 = blockIdx.y * BM;
  const int n0 = blockIdx.x * BN;
  const int half = lane >> 4;
  const int lr   = lane & 15;

  v8f acc[2][4] = {};

#if USE_TDM
  __shared__ unsigned short As[2][BM * LDT];
  __shared__ unsigned short Bs[2][BN * LDT];
  // Force a scalar branch: TDM ignores EXEC, so a vector-masked path would
  // issue duplicate DMAs from every wave.
  const int wid = __builtin_amdgcn_readfirstlane(wave);
  if (wid == 0) {
    tdm_load_tile((unsigned)(uintptr_t)&As[0][0], Xq + (size_t)m0 * IN_F,
                  BK, BM, IN_F);
    tdm_load_tile((unsigned)(uintptr_t)&Bs[0][0], Wb + (size_t)n0 * IN_F,
                  BK, BN, IN_F);
  }
  int cur = 0;
  for (int kb = 0; kb < IN_F; kb += BK) {
    if (wid == 0) __builtin_amdgcn_s_wait_tensorcnt(0);  // chunk `cur` landed
    __syncthreads();   // publish chunk; all prior reads of other buffer done
    if (wid == 0 && (kb + BK) < IN_F) {
      tdm_load_tile((unsigned)(uintptr_t)&As[cur ^ 1][0],
                    Xq + (size_t)m0 * IN_F + kb + BK, BK, BM, IN_F);
      tdm_load_tile((unsigned)(uintptr_t)&Bs[cur ^ 1][0],
                    Wb + (size_t)n0 * IN_F + kb + BK, BK, BN, IN_F);
    }
    const unsigned short* Ab = &As[cur][0];
    const unsigned short* Bb = &Bs[cur][0];
#pragma unroll
    for (int kk = 0; kk < BK; kk += 32) {
      v16bf af[2], bfr[4];
#pragma unroll
      for (int i = 0; i < 2; ++i)
        af[i] = load_frag(Ab + (wrow * 32 + i * 16 + lr) * LDT + kk + half * 8);
#pragma unroll
      for (int j = 0; j < 4; ++j)
        bfr[j] = load_frag(Bb + (wcol * 64 + j * 16 + lr) * LDT + kk + half * 8);
#pragma unroll
      for (int i = 0; i < 2; ++i)
#pragma unroll
        for (int j = 0; j < 4; ++j)
          acc[i][j] = __builtin_amdgcn_wmma_f32_16x16x32_bf16(
              false, af[i], false, bfr[j], (short)0, acc[i][j], false, false);
    }
    cur ^= 1;
  }
#else  // fallback: manual VGPR-staged single-buffer path
  __shared__ unsigned short As[BM * LDT];
  __shared__ unsigned short Bs[BN * LDT];
  for (int kb = 0; kb < IN_F; kb += BK) {
#pragma unroll
    for (int t = 0; t < 4; ++t) {
      int chunk = tid + t * 256;
      int row = chunk >> 3;
      int col = (chunk & 7) * 8;
      *(uint4*)(&As[row * LDT + col]) =
          *(const uint4*)(&Xq[(size_t)(m0 + row) * IN_F + kb + col]);
      *(uint4*)(&Bs[row * LDT + col]) =
          *(const uint4*)(&Wb[(size_t)(n0 + row) * IN_F + kb + col]);
    }
    __syncthreads();
#pragma unroll
    for (int kk = 0; kk < BK; kk += 32) {
      v16bf af[2], bfr[4];
#pragma unroll
      for (int i = 0; i < 2; ++i)
        af[i] = load_frag(&As[(wrow * 32 + i * 16 + lr) * LDT + kk + half * 8]);
#pragma unroll
      for (int j = 0; j < 4; ++j)
        bfr[j] = load_frag(&Bs[(wcol * 64 + j * 16 + lr) * LDT + kk + half * 8]);
#pragma unroll
      for (int i = 0; i < 2; ++i)
#pragma unroll
        for (int j = 0; j < 4; ++j)
          acc[i][j] = __builtin_amdgcn_wmma_f32_16x16x32_bf16(
              false, af[i], false, bfr[j], (short)0, acc[i][j], false, false);
    }
    __syncthreads();
  }
#endif

  // Epilogue: out = acc + 2*(T @ loraB) + bias
  int gn[4]; float bsv[4]; float lb[4][RANK];
#pragma unroll
  for (int j = 0; j < 4; ++j) {
    gn[j] = n0 + wcol * 64 + j * 16 + lr;
    bsv[j] = bias[gn[j]];
#pragma unroll
    for (int r = 0; r < RANK; ++r) lb[j][r] = loraB[(size_t)r * OUT_F + gn[j]];
  }
#pragma unroll
  for (int i = 0; i < 2; ++i) {
    int mbase = m0 + wrow * 32 + i * 16 + half * 8;  // C/D layout: VGPR v -> row
#pragma unroll
    for (int v = 0; v < 8; ++v) {
      int m = mbase + v;
      const float4* tp = (const float4*)(T + (size_t)m * RANK);
      float4 ta = tp[0], tb = tp[1];
#pragma unroll
      for (int j = 0; j < 4; ++j) {
        float d = ta.x * lb[j][0] + ta.y * lb[j][1] + ta.z * lb[j][2] + ta.w * lb[j][3]
                + tb.x * lb[j][4] + tb.y * lb[j][5] + tb.z * lb[j][6] + tb.w * lb[j][7];
        out[(size_t)m * OUT_F + gn[j]] = acc[i][j][v] + SCALING * d + bsv[j];
      }
    }
  }
}

// ---------------------------------------------------------------------------
extern "C" void kernel_launch(void* const* d_in, const int* in_sizes, int n_in,
                              void* d_out, int out_size, void* d_ws, size_t ws_size,
                              hipStream_t stream) {
  (void)in_sizes; (void)n_in; (void)out_size; (void)ws_size;
  const float* x       = (const float*)d_in[0];
  const float* s_vec   = (const float*)d_in[1];
  const float* a_scale = (const float*)d_in[2];
  const float* w_deq   = (const float*)d_in[3];
  const float* loraA   = (const float*)d_in[4];
  const float* loraB   = (const float*)d_in[5];
  const float* bias    = (const float*)d_in[6];
  float* out = (float*)d_out;

  // workspace layout (256B-aligned):
  //   a_vec : 4096 f32        (16 KB)
  //   T     : 8192*8 f32      (256 KB)
  //   Xq    : 8192*4096 bf16  (64 MB)
  //   Wb    : 4096*4096 bf16  (32 MB)
  char* ws = (char*)d_ws;
  float* a_vec = (float*)ws;
  float* T     = (float*)(ws + (16 << 10));
  unsigned short* Xq = (unsigned short*)(ws + (16 << 10) + (256 << 10));
  unsigned short* Wb = (unsigned short*)(ws + (16 << 10) + (256 << 10)
                                         + (size_t)MROWS * IN_F * 2);

  quantile_scale_kernel<<<IN_F / 256, 256, 0, stream>>>(x, s_vec, a_scale, a_vec);
  act_quant_kernel<<<(int)(((size_t)MROWS * IN_F / 4) / 256), 256, 0, stream>>>(
      x, s_vec, a_vec, Xq);
  wconv_kernel<<<(int)(((size_t)OUT_F * IN_F / 4) / 256), 256, 0, stream>>>(w_deq, Wb);
  lora_t_kernel<<<MROWS, 256, 0, stream>>>(Xq, loraA, T);
  dim3 grid(OUT_F / BN, MROWS / BM);
  wmma_gemm_kernel<<<grid, 256, 0, stream>>>(Xq, Wb, T, loraB, bias, out);
}
